// MixtureOfExperts_88837103551402
// MI455X (gfx1250) — matrix operand order repaired and checked by reading, not compiled, not run on hardware
//
#include <hip/hip_runtime.h>

// ---------------- problem constants ----------------
#define D_MODEL   1024
#define D_HIDDEN  4096
#define D_OUTW    1024
#define N_EXPERTS 8
#define TOKENS    8192      // 4 * 2048
#define LB_W      0.01f
#define KC        128       // K-chunk staged in LDS per double-buffer step
#define APAD      8         // bf16 pad per LDS row (16B) -> conflict-free ds_load_b128

typedef __attribute__((ext_vector_type(16))) __bf16 v16bf;
typedef __attribute__((ext_vector_type(8)))  __bf16 v8bf;
typedef __attribute__((ext_vector_type(4)))  __bf16 v4bf;
typedef __attribute__((ext_vector_type(8)))  float  v8f;

static __device__ __forceinline__ v16bf cat8(v8bf lo, v8bf hi) {
    return __builtin_shufflevector(lo, hi, 0,1,2,3,4,5,6,7,8,9,10,11,12,13,14,15);
}

static __device__ __forceinline__ v8f wmma_bf16(v16bf a, v16bf b, v8f c) {
    return __builtin_amdgcn_wmma_f32_16x16x32_bf16(false, a, false, b, (short)0, c, false, false);
}

// async copy 16B from global to LDS (per-lane addresses), tracked by ASYNCcnt
static __device__ __forceinline__ void async_b128(unsigned lds_off, const __bf16* gptr) {
    asm volatile("global_load_async_to_lds_b128 %0, %1, off"
                 :: "v"(lds_off), "v"((unsigned long long)(uintptr_t)gptr)
                 : "memory");
}
static __device__ __forceinline__ void wait_async0() {
    asm volatile("s_wait_asynccnt 0" ::: "memory");
}

// ---------------- zero output + counters ----------------
__global__ void moe_zero(float* __restrict__ out, int n, int* __restrict__ cnt) {
    int i = blockIdx.x * blockDim.x + threadIdx.x;
    if (i < n) out[i] = 0.0f;
    if (i < N_EXPERTS) cnt[i] = 0;
}

// ---------------- f32 -> bf16 conversion (vectorized) ----------------
__global__ void moe_cvt(const float* __restrict__ s, __bf16* __restrict__ d, int n) {
    int i = (blockIdx.x * blockDim.x + threadIdx.x) * 4;
    if (i + 3 < n) {
        float4 v = *(const float4*)(s + i);
        v4bf o;
        o[0] = (__bf16)v.x; o[1] = (__bf16)v.y; o[2] = (__bf16)v.z; o[3] = (__bf16)v.w;
        *(v4bf*)(d + i) = o;
    }
}

// ---------------- router: logits, softmax, top-2, gather lists ----------------
__global__ void moe_router(const float* __restrict__ x, const float* __restrict__ rw,
                           const float* __restrict__ rb,
                           float* __restrict__ probs, int* __restrict__ cnt,
                           int* __restrict__ etok, float* __restrict__ ewt) {
    int t = blockIdx.x * blockDim.x + threadIdx.x;
    if (t >= TOKENS) return;

    float acc[N_EXPERTS];
#pragma unroll
    for (int e = 0; e < N_EXPERTS; ++e) acc[e] = 0.0f;

    const float4* xr = (const float4*)(x + (size_t)t * D_MODEL);
    const float4* wr = (const float4*)rw;
    for (int d = 0; d < D_MODEL / 4; ++d) {
        float4 xv = xr[d];
#pragma unroll
        for (int e = 0; e < N_EXPERTS; ++e) {
            float4 wv = wr[e * (D_MODEL / 4) + d];
            acc[e] = fmaf(xv.x, wv.x, fmaf(xv.y, wv.y, fmaf(xv.z, wv.z, fmaf(xv.w, wv.w, acc[e]))));
        }
    }

    float mx = -1e30f;
#pragma unroll
    for (int e = 0; e < N_EXPERTS; ++e) { acc[e] += rb[e]; mx = fmaxf(mx, acc[e]); }
    float p[N_EXPERTS], sum = 0.0f;
#pragma unroll
    for (int e = 0; e < N_EXPERTS; ++e) { p[e] = __expf(acc[e] - mx); sum += p[e]; }
    float inv = 1.0f / sum;
#pragma unroll
    for (int e = 0; e < N_EXPERTS; ++e) { p[e] *= inv; probs[(size_t)t * N_EXPERTS + e] = p[e]; }

    int i1 = 0;
#pragma unroll
    for (int e = 1; e < N_EXPERTS; ++e) if (p[e] > p[i1]) i1 = e;
    int i2 = (i1 == 0) ? 1 : 0;
#pragma unroll
    for (int e = 0; e < N_EXPERTS; ++e) if (e != i1 && p[e] > p[i2]) i2 = e;

    float pa = p[i1], pb = p[i2];
    float invs = 1.0f / (pa + pb);
    int s1 = atomicAdd(&cnt[i1], 1);
    etok[i1 * TOKENS + s1] = t;  ewt[i1 * TOKENS + s1] = pa * invs;
    int s2 = atomicAdd(&cnt[i2], 1);
    etok[i2 * TOKENS + s2] = t;  ewt[i2 * TOKENS + s2] = pb * invs;
}

// ---------------- tiny exclusive scan over 8 expert counts ----------------
__global__ void moe_scan(const int* __restrict__ cnt, int* __restrict__ eoff) {
    if (threadIdx.x == 0) {
        int s = 0;
#pragma unroll
        for (int e = 0; e < N_EXPERTS; ++e) { eoff[e] = s; s += cnt[e]; }
    }
}

// =====================================================================
// GEMM core: block = 256 threads = 8 waves arranged 2(M) x 4(N).
// Block tile: 64 rows x 128 cols. Wave tile: 32x32 (4 accumulators).
// A (64 x KC) staged in LDS via async copies, double buffered.
// =====================================================================

// ---------------- GEMM1: h = relu(gather(x) @ w1[e]^T + b1[e]) ----------------
__global__ void __launch_bounds__(256)
moe_gemm1(const __bf16* __restrict__ xb, const __bf16* __restrict__ w1b,
          const float* __restrict__ b1,
          const int* __restrict__ cnt, const int* __restrict__ eoff,
          const int* __restrict__ etok, __bf16* __restrict__ hb) {
    const int e  = blockIdx.z;
    const int ce = cnt[e];
    const int mb = blockIdx.y * 64;
    if (mb >= ce) return;

    const int tid  = threadIdx.x;
    const int wave = tid >> 5, lane = tid & 31;
    const int half = lane >> 4, l15 = lane & 15;
    const int wm = wave >> 2, wn = wave & 3;
    const int mw = mb + wm * 32;                    // wave's first slot row
    const int nw = blockIdx.x * 128 + wn * 32;      // wave's first column

    __shared__ __bf16 sA[2][64][KC + APAD];
    const unsigned ldsbase   = (unsigned)(uintptr_t)(void*)&sA[0][0][0];
    const unsigned bufstride = (unsigned)(64 * (KC + APAD) * 2);

    // per-thread staging: 4 x 16B per chunk (256 thr * 16B * 4 = 16KB = 64*KC bf16)
    const __bf16* gsrc[4];
    unsigned      ldst[4];
#pragma unroll
    for (int it = 0; it < 4; ++it) {
        int i  = tid + 256 * it;
        int r  = i >> 4;               // staging row 0..63
        int kk = (i & 15) * 8;         // bf16 offset in row
        int slot = mb + r; if (slot >= ce) slot = ce - 1;
        int tok  = etok[e * TOKENS + slot];
        gsrc[it] = xb + (size_t)tok * D_MODEL + kk;
        ldst[it] = ldsbase + (unsigned)((r * (KC + APAD) + kk) * 2);
    }

    const __bf16* B0 = w1b + ((size_t)e * D_HIDDEN + nw      + l15) * D_MODEL + 16 * half;
    const __bf16* B1 = w1b + ((size_t)e * D_HIDDEN + nw + 16 + l15) * D_MODEL + 16 * half;

    v8f c00 = {0.f,0.f,0.f,0.f,0.f,0.f,0.f,0.f};
    v8f c01 = c00, c10 = c00, c11 = c00;

    // prologue: chunk 0 -> buffer 0
#pragma unroll
    for (int it = 0; it < 4; ++it) async_b128(ldst[it], gsrc[it]);

    const int NCH = D_MODEL / KC;
    for (int ch = 0; ch < NCH; ++ch) {
        wait_async0();
        __syncthreads();
        const int cur = ch & 1;
        if (ch + 1 < NCH) {
            const unsigned nb = (unsigned)(cur ^ 1) * bufstride;
#pragma unroll
            for (int it = 0; it < 4; ++it)
                async_b128(ldst[it] + nb, gsrc[it] + (ch + 1) * KC);
        }
        const int kc = ch * KC;
        __builtin_prefetch(B0 + kc + KC, 0, 1);
        __builtin_prefetch(B1 + kc + KC, 0, 1);
#pragma unroll
        for (int k0 = 0; k0 < KC; k0 += 32) {
            const __bf16* a0 = &sA[cur][wm * 32      + l15][k0 + 16 * half];
            const __bf16* a1 = &sA[cur][wm * 32 + 16 + l15][k0 + 16 * half];
            v16bf A0 = cat8(*(const v8bf*)a0, *(const v8bf*)(a0 + 8));
            v16bf A1 = cat8(*(const v8bf*)a1, *(const v8bf*)(a1 + 8));
            v16bf Bv0 = cat8(*(const v8bf*)(B0 + kc + k0), *(const v8bf*)(B0 + kc + k0 + 8));
            v16bf Bv1 = cat8(*(const v8bf*)(B1 + kc + k0), *(const v8bf*)(B1 + kc + k0 + 8));
            c00 = wmma_bf16(A0, Bv0, c00);
            c01 = wmma_bf16(A0, Bv1, c01);
            c10 = wmma_bf16(A1, Bv0, c10);
            c11 = wmma_bf16(A1, Bv1, c11);
        }
    }

    const int   hbase = eoff[e];
    const float bias0 = b1[e * D_HIDDEN + nw      + l15];
    const float bias1 = b1[e * D_HIDDEN + nw + 16 + l15];
#pragma unroll
    for (int mt = 0; mt < 2; ++mt) {
        v8f cm0 = mt ? c10 : c00;
        v8f cm1 = mt ? c11 : c01;
#pragma unroll
        for (int i = 0; i < 8; ++i) {
            int slot = mw + mt * 16 + i + 8 * half;
            if (slot < ce) {
                size_t rowb = (size_t)(hbase + slot) * D_HIDDEN;
                float v0 = cm0[i] + bias0; v0 = v0 > 0.0f ? v0 : 0.0f;
                float v1 = cm1[i] + bias1; v1 = v1 > 0.0f ? v1 : 0.0f;
                hb[rowb + nw      + l15] = (__bf16)v0;
                hb[rowb + nw + 16 + l15] = (__bf16)v1;
            }
        }
    }
}

// ---------------- GEMM2: out[tok] += wt * (h @ w2[e]^T + b2[e]) ----------------
__global__ void __launch_bounds__(256)
moe_gemm2(const __bf16* __restrict__ hb, const __bf16* __restrict__ w2b,
          const float* __restrict__ b2,
          const int* __restrict__ cnt, const int* __restrict__ eoff,
          const int* __restrict__ etok, const float* __restrict__ ewt,
          float* __restrict__ out) {
    const int e  = blockIdx.z;
    const int ce = cnt[e];
    const int mb = blockIdx.y * 64;
    if (mb >= ce) return;

    const int tid  = threadIdx.x;
    const int wave = tid >> 5, lane = tid & 31;
    const int half = lane >> 4, l15 = lane & 15;
    const int wm = wave >> 2, wn = wave & 3;
    const int mw = mb + wm * 32;
    const int nw = blockIdx.x * 128 + wn * 32;
    const int hbase = eoff[e];

    __shared__ __bf16 sA[2][64][KC + APAD];
    const unsigned ldsbase   = (unsigned)(uintptr_t)(void*)&sA[0][0][0];
    const unsigned bufstride = (unsigned)(64 * (KC + APAD) * 2);

    const __bf16* gsrc[4];
    unsigned      ldst[4];
#pragma unroll
    for (int it = 0; it < 4; ++it) {
        int i  = tid + 256 * it;
        int r  = i >> 4;
        int kk = (i & 15) * 8;
        int slot = mb + r; if (slot >= ce) slot = ce - 1;
        gsrc[it] = hb + (size_t)(hbase + slot) * D_HIDDEN + kk;
        ldst[it] = ldsbase + (unsigned)((r * (KC + APAD) + kk) * 2);
    }

    const __bf16* B0 = w2b + ((size_t)e * D_OUTW + nw      + l15) * D_HIDDEN + 16 * half;
    const __bf16* B1 = w2b + ((size_t)e * D_OUTW + nw + 16 + l15) * D_HIDDEN + 16 * half;

    v8f c00 = {0.f,0.f,0.f,0.f,0.f,0.f,0.f,0.f};
    v8f c01 = c00, c10 = c00, c11 = c00;

#pragma unroll
    for (int it = 0; it < 4; ++it) async_b128(ldst[it], gsrc[it]);

    const int NCH = D_HIDDEN / KC;
    for (int ch = 0; ch < NCH; ++ch) {
        wait_async0();
        __syncthreads();
        const int cur = ch & 1;
        if (ch + 1 < NCH) {
            const unsigned nb = (unsigned)(cur ^ 1) * bufstride;
#pragma unroll
            for (int it = 0; it < 4; ++it)
                async_b128(ldst[it] + nb, gsrc[it] + (ch + 1) * KC);
        }
        const int kc = ch * KC;
        __builtin_prefetch(B0 + kc + KC, 0, 1);
        __builtin_prefetch(B1 + kc + KC, 0, 1);
#pragma unroll
        for (int k0 = 0; k0 < KC; k0 += 32) {
            const __bf16* a0 = &sA[cur][wm * 32      + l15][k0 + 16 * half];
            const __bf16* a1 = &sA[cur][wm * 32 + 16 + l15][k0 + 16 * half];
            v16bf A0 = cat8(*(const v8bf*)a0, *(const v8bf*)(a0 + 8));
            v16bf A1 = cat8(*(const v8bf*)a1, *(const v8bf*)(a1 + 8));
            v16bf Bv0 = cat8(*(const v8bf*)(B0 + kc + k0), *(const v8bf*)(B0 + kc + k0 + 8));
            v16bf Bv1 = cat8(*(const v8bf*)(B1 + kc + k0), *(const v8bf*)(B1 + kc + k0 + 8));
            c00 = wmma_bf16(A0, Bv0, c00);
            c01 = wmma_bf16(A0, Bv1, c01);
            c10 = wmma_bf16(A1, Bv0, c10);
            c11 = wmma_bf16(A1, Bv1, c11);
        }
    }

    const float bias0 = b2[e * D_OUTW + nw      + l15];
    const float bias1 = b2[e * D_OUTW + nw + 16 + l15];
#pragma unroll
    for (int mt = 0; mt < 2; ++mt) {
        v8f cm0 = mt ? c10 : c00;
        v8f cm1 = mt ? c11 : c01;
#pragma unroll
        for (int i = 0; i < 8; ++i) {
            int slot = mw + mt * 16 + i + 8 * half;
            if (slot < ce) {
                int   tok = etok[e * TOKENS + slot];
                float wt  = ewt[e * TOKENS + slot];
                float v0  = (cm0[i] + bias0) * wt;
                float v1  = (cm1[i] + bias1) * wt;
                // each output element receives exactly 2 adds (top-2) -> commutative, deterministic
                unsafeAtomicAdd(out + (size_t)tok * D_OUTW + nw      + l15, v0);
                unsafeAtomicAdd(out + (size_t)tok * D_OUTW + nw + 16 + l15, v1);
            }
        }
    }
}

// ---------------- load-balance loss (deterministic fixed-order reduction) ----------------
__global__ void moe_finalize(const float* __restrict__ probs, const int* __restrict__ cnt,
                             float* __restrict__ out_loss) {
    __shared__ float red[256];
    float psum[N_EXPERTS];
#pragma unroll
    for (int e = 0; e < N_EXPERTS; ++e) psum[e] = 0.0f;
    for (int t = threadIdx.x; t < TOKENS; t += 256) {
#pragma unroll
        for (int e = 0; e < N_EXPERTS; ++e) psum[e] += probs[(size_t)t * N_EXPERTS + e];
    }
    float loss = 0.0f;
    for (int e = 0; e < N_EXPERTS; ++e) {
        red[threadIdx.x] = psum[e];
        __syncthreads();
        for (int s = 128; s > 0; s >>= 1) {
            if (threadIdx.x < s) red[threadIdx.x] += red[threadIdx.x + s];
            __syncthreads();
        }
        if (threadIdx.x == 0) {
            float mean_prob  = red[0] / (float)TOKENS;
            float usage_frac = (float)cnt[e] / (float)(TOKENS * 2);
            loss += usage_frac * mean_prob;
        }
        __syncthreads();
    }
    if (threadIdx.x == 0) out_loss[0] = LB_W * (float)N_EXPERTS * loss;
}

// ---------------- launch ----------------
extern "C" void kernel_launch(void* const* d_in, const int* in_sizes, int n_in,
                              void* d_out, int out_size, void* d_ws, size_t ws_size,
                              hipStream_t stream) {
    const float* x  = (const float*)d_in[0];
    const float* rw = (const float*)d_in[1];
    const float* rb = (const float*)d_in[2];
    const float* w1 = (const float*)d_in[3];
    const float* b1 = (const float*)d_in[4];
    const float* w2 = (const float*)d_in[5];
    const float* b2 = (const float*)d_in[6];
    float* out = (float*)d_out;

    char* ws = (char*)d_ws;
    size_t o = 0;
    int*    cnt   = (int*)(ws + o);   o += N_EXPERTS * sizeof(int);
    int*    eoff  = (int*)(ws + o);   o += N_EXPERTS * sizeof(int);
    o = (o + 255) & ~(size_t)255;
    int*    etok  = (int*)(ws + o);   o += (size_t)N_EXPERTS * TOKENS * sizeof(int);
    float*  ewt   = (float*)(ws + o); o += (size_t)N_EXPERTS * TOKENS * sizeof(float);
    float*  probs = (float*)(ws + o); o += (size_t)TOKENS * N_EXPERTS * sizeof(float);
    o = (o + 255) & ~(size_t)255;
    __bf16* xb    = (__bf16*)(ws + o); o += (size_t)TOKENS * D_MODEL * 2;
    __bf16* w1b   = (__bf16*)(ws + o); o += (size_t)N_EXPERTS * D_HIDDEN * D_MODEL * 2;
    __bf16* w2b   = (__bf16*)(ws + o); o += (size_t)N_EXPERTS * D_OUTW * D_HIDDEN * 2;
    __bf16* hb    = (__bf16*)(ws + o); o += (size_t)2 * TOKENS * D_HIDDEN * 2;

    const int outN = TOKENS * D_OUTW + 1;
    moe_zero<<<(outN + 255) / 256, 256, 0, stream>>>(out, outN, cnt);
    moe_router<<<TOKENS / 256, 256, 0, stream>>>(x, rw, rb, probs, cnt, etok, ewt);
    moe_scan<<<1, 32, 0, stream>>>(cnt, eoff);

    const int nx  = TOKENS * D_MODEL;
    const int nw1 = N_EXPERTS * D_HIDDEN * D_MODEL;
    const int nw2 = N_EXPERTS * D_OUTW * D_HIDDEN;
    moe_cvt<<<(nx  / 4 + 255) / 256, 256, 0, stream>>>(x,  xb,  nx);
    moe_cvt<<<(nw1 / 4 + 255) / 256, 256, 0, stream>>>(w1, w1b, nw1);
    moe_cvt<<<(nw2 / 4 + 255) / 256, 256, 0, stream>>>(w2, w2b, nw2);

    moe_gemm1<<<dim3(D_HIDDEN / 128, TOKENS / 64, N_EXPERTS), 256, 0, stream>>>(
        xb, w1b, b1, cnt, eoff, etok, hb);
    moe_gemm2<<<dim3(D_OUTW / 128, TOKENS / 64, N_EXPERTS), 256, 0, stream>>>(
        hb, w2b, b2, cnt, eoff, etok, ewt, out);

    moe_finalize<<<1, 256, 0, stream>>>(probs, cnt, out + (size_t)TOKENS * D_OUTW);
}
